// SparseAttention_67757404062393
// MI455X (gfx1250) — matrix operand order repaired and checked by reading, not compiled, not run on hardware
//
#include <hip/hip_runtime.h>
#include <hip/hip_bf16.h>

typedef __attribute__((ext_vector_type(16))) _Float16 v16h;
typedef __attribute__((ext_vector_type(8)))  _Float16 v8h;
typedef __attribute__((ext_vector_type(4)))  _Float16 v4h;
typedef __attribute__((ext_vector_type(8)))  float    v8f;

// Problem constants (from reference): B=4, S=4096, E=1024, H=8, d=128, bs=128
static constexpr int Bn  = 4;
static constexpr int Sn  = 4096;
static constexpr int En  = 1024;
static constexpr int Hn  = 8;
static constexpr int Dn  = 128;   // head dim
static constexpr int BS  = 128;   // block size (rows per attention block)
static constexpr int NB  = Sn / BS;  // 32 blocks along sequence
static constexpr int PH  = 136;   // LDS pitch in halfs (272B, 16B aligned, avoids bank conflicts)

#define FULL16(lo, hi) __builtin_shufflevector(lo, hi, 0,1,2,3,4,5,6,7,8,9,10,11,12,13,14,15)

// One wave computes a 16x128 slab of C = A(16x128) * B^T-ish(128x128) using
// v_wmma_f32_16x16x32_f16. A is row-major [128][PH] halfs; Bm is row-major with
// row n holding the 128 K-values of output column n (i.e. K rows for QK^T,
// V^T rows for PV). acc[t] is the 16x16 tile at columns t*16..t*16+15.
__device__ __forceinline__ void mma_slab(const _Float16* __restrict__ A, int row0,
                                         const _Float16* __restrict__ Bm,
                                         v8f acc[8], int lane) {
    const int m    = lane & 15;
    const int half = lane >> 4;
#pragma unroll
    for (int kk = 0; kk < 4; ++kk) {
        const int ka = kk * 32 + half * 8;      // A: VGPR0-3 k=8*half.., VGPR4-7 k=16+8*half..
        v8h alo = *(const v8h*)&A[(row0 + m) * PH + ka];
        v8h ahi = *(const v8h*)&A[(row0 + m) * PH + ka + 16];
        v16h av = FULL16(alo, ahi);
        const int kb = kk * 32 + half * 8;      // B: same k order as A
#pragma unroll
        for (int t = 0; t < 8; ++t) {
            const int n = t * 16 + m;
            v8h blo = *(const v8h*)&Bm[n * PH + kb];
            v8h bhi = *(const v8h*)&Bm[n * PH + kb + 16];
            v16h bv = FULL16(blo, bhi);
            acc[t] = __builtin_amdgcn_wmma_f32_16x16x32_f16(
                false, av, false, bv, (short)0, acc[t], false, false);
        }
    }
}

__global__ void block_attn_kernel(const float* __restrict__ q,
                                  const float* __restrict__ k,
                                  const float* __restrict__ v,
                                  float* __restrict__ out) {
    // LDS: 3 tiles of 128 x PH f16 = 3 * 34816 B = 104448 B. sP aliases sQ (Q dead after GEMM1).
    __shared__ __align__(16) _Float16 sQ[BS * PH];
    __shared__ __align__(16) _Float16 sK[BS * PH];
    __shared__ __align__(16) _Float16 sVt[BS * PH];  // V transposed: row n = head-dim col n
    _Float16* sP = sQ;

    const int bid = blockIdx.x;
    const int n   = bid & (NB - 1);
    const int h   = (bid >> 5) & (Hn - 1);
    const int b   = bid >> 8;
    const int s0  = n * BS;
    const int tid = threadIdx.x;

    // ---- Stage Q, K (row-major f16) and V (transposed f16) into LDS ----
    {
        const int row = tid >> 1;            // 0..127
        const int cb  = (tid & 1) * 64;      // column half
        const size_t gbase = ((size_t)(b * Sn + s0 + row)) * En + h * Dn + cb;
        const float4* gq = (const float4*)(q + gbase);
        const float4* gk = (const float4*)(k + gbase);
        const float4* gv = (const float4*)(v + gbase);
#pragma unroll
        for (int i = 0; i < 16; ++i) {
            float4 fq = gq[i];
            float4 fk = gk[i];
            float4 fv = gv[i];
            v4h hq = {(_Float16)fq.x, (_Float16)fq.y, (_Float16)fq.z, (_Float16)fq.w};
            v4h hk = {(_Float16)fk.x, (_Float16)fk.y, (_Float16)fk.z, (_Float16)fk.w};
            *(v4h*)&sQ[row * PH + cb + i * 4] = hq;
            *(v4h*)&sK[row * PH + cb + i * 4] = hk;
            const int c = cb + i * 4;
            sVt[(c + 0) * PH + row] = (_Float16)fv.x;
            sVt[(c + 1) * PH + row] = (_Float16)fv.y;
            sVt[(c + 2) * PH + row] = (_Float16)fv.z;
            sVt[(c + 3) * PH + row] = (_Float16)fv.w;
        }
    }
    __syncthreads();

    const int lane = tid & 31;
    const int wv   = tid >> 5;        // 8 waves, wave wv owns rows wv*16 .. wv*16+15
    const int row0 = wv * 16;
    const int m16  = lane & 15;
    const int half = lane >> 4;

    // ---- GEMM1: S = Q * K^T (accumulate fp32 in registers) ----
    v8f accS[8];
#pragma unroll
    for (int t = 0; t < 8; ++t) accS[t] = (v8f)(0.0f);
    mma_slab(sQ, row0, sK, accS, lane);

    __syncthreads();  // all Q reads done; sQ may be overwritten by P

    // ---- Softmax over each row (128 cols) in registers ----
    // C layout: accS[t][r] @ lane -> row = row0 + 8*half + r, col = t*16 + m16.
    // A full row lives in one 16-lane group -> shfl_xor masks 1,2,4,8.
    const float scale = 0.08838834764831845f;  // 1/sqrt(128)
#pragma unroll
    for (int r = 0; r < 8; ++r) {
        float mx = -3.0e38f;
#pragma unroll
        for (int t = 0; t < 8; ++t) {
            accS[t][r] *= scale;
            mx = fmaxf(mx, accS[t][r]);
        }
#pragma unroll
        for (int off = 1; off < 16; off <<= 1) mx = fmaxf(mx, __shfl_xor(mx, off));
        float sum = 0.0f;
#pragma unroll
        for (int t = 0; t < 8; ++t) {
            float e = __expf(accS[t][r] - mx);
            accS[t][r] = e;
            sum += e;
        }
#pragma unroll
        for (int off = 1; off < 16; off <<= 1) sum += __shfl_xor(sum, off);
        const float inv = 1.0f / sum;
#pragma unroll
        for (int t = 0; t < 8; ++t) {
            sP[(row0 + 8 * half + r) * PH + t * 16 + m16] =
                (_Float16)(accS[t][r] * inv);
        }
    }
    __syncthreads();  // P fully written before GEMM2 reads it

    // ---- GEMM2: O = P * V  (B operand from V^T rows) ----
    v8f accO[8];
#pragma unroll
    for (int t = 0; t < 8; ++t) accO[t] = (v8f)(0.0f);
    mma_slab(sP, row0, sVt, accO, lane);

    // ---- Store O (fp32, coalesced across 16-lane groups) ----
#pragma unroll
    for (int t = 0; t < 8; ++t) {
#pragma unroll
        for (int r = 0; r < 8; ++r) {
            const int orow = row0 + 8 * half + r;
            out[((size_t)(b * Sn + s0 + orow)) * En + h * Dn + t * 16 + m16] =
                accO[t][r];
        }
    }
}

extern "C" void kernel_launch(void* const* d_in, const int* in_sizes, int n_in,
                              void* d_out, int out_size, void* d_ws, size_t ws_size,
                              hipStream_t stream) {
    const float* q = (const float*)d_in[0];
    const float* k = (const float*)d_in[1];
    const float* v = (const float*)d_in[2];
    float* out = (float*)d_out;
    (void)in_sizes; (void)n_in; (void)out_size; (void)d_ws; (void)ws_size;
    dim3 grid(Bn * Hn * NB);   // 1024 independent attention blocks
    dim3 block(256);           // 8 wave32
    block_attn_kernel<<<grid, block, 0, stream>>>(q, k, v, out);
}
